// QuantizedLinear_18021682774281
// MI455X (gfx1250) — compile-verified, matrix operand order checked by reading, not run on hardware
//
#include <hip/hip_runtime.h>
#include <hip/hip_bf16.h>

#define IN_FEATURES  4096
#define OUT_FEATURES 11008
#define BATCH        8
#define GROUP_SIZE   128
#define NUM_GROUPS   32          // IN/GROUP
#define PACKED_PER_ROW (IN_FEATURES / 2)   // int32 elements per weight row (1 byte payload each)

typedef __attribute__((ext_vector_type(16))) _Float16 v16h;
typedef __attribute__((ext_vector_type(8)))  float    v8f;
typedef __attribute__((ext_vector_type(4)))  unsigned int u32x4;
typedef __attribute__((ext_vector_type(4)))  float    f32x4;

union AFrag { u32x4 q[2]; v16h h; };
union BFrag { unsigned int u[8]; v16h h; };

// ---------------------------------------------------------------------------
// Prep kernel: x f32 [8,4096] -> x16 f16 [16,4096] (rows 8..15 zero),
// plus per-group column sums xsum[g][r] (g in 0..31, r in 0..7).
// ---------------------------------------------------------------------------
__global__ void ql_prep_kernel(const float* __restrict__ x,
                               _Float16* __restrict__ x16,
                               float* __restrict__ xsum) {
    if (blockIdx.x < 64) {
        int idx = blockIdx.x * 256 + threadIdx.x;           // 16384 threads
        #pragma unroll
        for (int i = 0; i < 4; ++i) {
            int e = idx + i * 16384;                        // 0 .. 65535
            int r = e >> 12;                                // row 0..15
            int k = e & 4095;
            float v = (r < BATCH) ? x[(r << 12) + k] : 0.0f;
            x16[e] = (_Float16)v;
        }
    } else {
        // 256 threads: t -> (group g = t>>3, batch row r = t&7)
        int t = threadIdx.x;
        int g = t >> 3;
        int r = t & 7;
        const float* p = x + (size_t)r * IN_FEATURES + g * GROUP_SIZE;
        float s = 0.0f;
        #pragma unroll 8
        for (int j = 0; j < GROUP_SIZE; ++j) s += p[j];
        xsum[g * 8 + r] = s;
    }
}

// ---------------------------------------------------------------------------
// Main kernel: one block (256 thr = 8 waves) per 16-column output tile.
// Wave w handles groups [4w, 4w+4); per group, 4x WMMA f32_16x16x32_f16
// accumulate Sum((q+1024)*x); then fp32 per-group scale + folded zero term.
// LDS reduction across the 8 waves; lanes 0..15 store rows 0..7 + bias.
// ---------------------------------------------------------------------------
__global__ void __launch_bounds__(256)
ql_main_kernel(const _Float16* __restrict__ x16,
               const float* __restrict__ xsum,
               const int* __restrict__ wq,
               const float* __restrict__ scales,
               const float* __restrict__ zeros,
               const float* __restrict__ bias,
               float* __restrict__ out) {
    __shared__ float red[8 * 256];   // [r][tid]

    const int tid  = threadIdx.x;
    const int wave = tid >> 5;
    const int lane = tid & 31;
    const int n    = lane & 15;          // output column within tile
    const int h    = lane >> 4;          // K-half selector per WMMA layout
    const int obase = blockIdx.x * 16;
    const int o    = obase + n;
    const float hmask = (h == 0) ? 1.0f : 0.0f;

    const unsigned* wrow = (const unsigned*)(wq) + (size_t)o * PACKED_PER_ROW;

    v8f acc = {};

    #pragma unroll
    for (int gi = 0; gi < 4; ++gi) {
        const int g = wave * 4 + gi;

        // prefetch next group's weight bytes for this lane (speculative; OOB dropped)
        __builtin_prefetch(wrow + (size_t)(g + 1) * (GROUP_SIZE / 2) + h * 4, 0, 1);

        v8f cg = {};
        #pragma unroll
        for (int c = 0; c < 4; ++c) {
            const int K0 = g * GROUP_SIZE + c * 32;         // base K of this 16x16x32 step

            // ---- A fragment: x16 row M=n, K runs {K0+8h..+7} and {K0+16+8h..+7}
            const _Float16* xr = x16 + (size_t)n * IN_FEATURES + K0 + h * 8;
            AFrag af;
            af.q[0] = *(const u32x4*)(xr);
            af.q[1] = *(const u32x4*)(xr + 16);

            // ---- B fragment: column o of W, same K runs. byte index = k/2.
            const unsigned* wp = wrow + (K0 >> 1) + h * 4;
            u32x4 w0 = *(const u32x4*)(wp);
            u32x4 w1 = *(const u32x4*)(wp + 8);

            // nibble -> f16 (value = 1024 + q), exact; low nibble = even K
            BFrag bf;
            #pragma unroll
            for (int j = 0; j < 4; ++j) {
                unsigned a = w0[j];
                unsigned b = w1[j];
                bf.u[j]     = 0x64006400u | (a & 0xFu) | ((a << 12) & 0xF0000u);
                bf.u[4 + j] = 0x64006400u | (b & 0xFu) | ((b << 12) & 0xF0000u);
            }

            cg = __builtin_amdgcn_wmma_f32_16x16x32_f16(
                     false, af.h, false, bf.h, (short)0, cg, false, false);
        }

        // fp32 group epilogue: scale*cg + (zero - 1024*scale) * groupsum(x)
        const float s  = scales[(size_t)o * NUM_GROUPS + g];
        const float z  = zeros [(size_t)o * NUM_GROUPS + g];
        const float zp = (z - 1024.0f * s) * hmask;
        const float* Sg = xsum + g * 8;
        f32x4 s0 = *(const f32x4*)(Sg);
        f32x4 s1 = *(const f32x4*)(Sg + 4);
        float sv[8] = { s0[0], s0[1], s0[2], s0[3], s1[0], s1[1], s1[2], s1[3] };
        #pragma unroll
        for (int r = 0; r < 8; ++r)
            acc[r] = fmaf(s, cg[r], fmaf(zp, sv[r], acc[r]));
    }

    // cross-wave reduction through LDS
    #pragma unroll
    for (int r = 0; r < 8; ++r) red[r * 256 + tid] = acc[r];
    __syncthreads();

    if (tid < 16) {                       // lanes holding batch rows 0..7
        const int oo = obase + tid;
        const float bb = bias[oo];
        #pragma unroll
        for (int r = 0; r < 8; ++r) {
            float t = 0.0f;
            #pragma unroll
            for (int w = 0; w < 8; ++w) t += red[r * 256 + w * 32 + tid];
            out[(size_t)r * OUT_FEATURES + oo] = t + bb;
        }
    }
}

// ---------------------------------------------------------------------------
extern "C" void kernel_launch(void* const* d_in, const int* in_sizes, int n_in,
                              void* d_out, int out_size, void* d_ws, size_t ws_size,
                              hipStream_t stream) {
    (void)in_sizes; (void)n_in; (void)out_size; (void)ws_size;

    const float* x       = (const float*)d_in[0];
    const int*   wq      = (const int*)  d_in[1];
    const float* scales  = (const float*)d_in[2];
    const float* zeros   = (const float*)d_in[3];
    const float* bias    = (const float*)d_in[4];
    float*       out     = (float*)d_out;

    // workspace: x16 f16 [16][4096] (128 KB) then xsum f32 [32][8] (1 KB)
    _Float16* x16  = (_Float16*)d_ws;
    float*    xsum = (float*)((char*)d_ws + (size_t)16 * IN_FEATURES * sizeof(_Float16));

    ql_prep_kernel<<<65, 256, 0, stream>>>(x, x16, xsum);
    ql_main_kernel<<<OUT_FEATURES / 16, 256, 0, stream>>>(
        x16, xsum, wq, scales, zeros, bias, out);
}
